// NSSM_29222957482913
// MI455X (gfx1250) — matrix operand order repaired and checked by reading, not compiled, not run on hardware
//
#include <hip/hip_runtime.h>

// CDNA5 / gfx1250, wave32. One wave runs the whole sequential scan.
// Hidden-unit "slot" numbering is chosen so the WMMA D-column layout of one
// step is exactly the B-operand layout of the next step -> no cross-lane ops.
typedef _Float16 v16h __attribute__((ext_vector_type(16)));
typedef _Float16 h2   __attribute__((ext_vector_type(2)));
typedef float    v8f  __attribute__((ext_vector_type(8)));
typedef int      v8i  __attribute__((ext_vector_type(8)));

static __device__ __forceinline__ h2 pk2(float a, float b) {
  return __builtin_bit_cast(h2, __builtin_amdgcn_cvt_pkrtz(a, b));
}

static __device__ __forceinline__ v8f wmma(v16h a, v16h b, v8f c) {
  // D = A(16x32 f16) x B(32x16 f16) + C(16x16 f32)
  return __builtin_amdgcn_wmma_f32_16x16x32_f16(false, a, false, b, (short)0, c, false, false);
}

// Hidden slot of (tile, rowM): tile0 row M -> M<8 ? M : M+8
//                              tile1 row M -> M<8 ? M+8 : M+16
// With this numbering, lane L's own acc regs (tile0 v0-7, tile1 v0-7) hold
// slots {0..15} (L<16) / {16..31} (L>=16) in order == B-operand K layout.
static __device__ __forceinline__ int slot0(int M) { return (M < 8) ? M : M + 8; }
static __device__ __forceinline__ int slot1(int M) { return (M < 8) ? M + 8 : M + 16; }

// f32->f16 pack + relu in the packed domain (cvt is monotone, no NaNs here,
// so relu(cvt(x)) == cvt(relu(x))): 8 v_cvt_pk_rtz + 8 v_pk_max per vector.
static __device__ __forceinline__ v16h marshal_relu(v8f t0, v8f t1) {
  const h2 zh = {};
  v8i b;
#pragma unroll
  for (int r = 0; r < 4; ++r) {
    const h2 x0 = pk2(t0[2*r], t0[2*r+1]);
    const h2 x1 = pk2(t1[2*r], t1[2*r+1]);
    b[r]     = __builtin_bit_cast(int, __builtin_elementwise_max(x0, zh));
    b[r + 4] = __builtin_bit_cast(int, __builtin_elementwise_max(x1, zh));
  }
  return __builtin_bit_cast(v16h, b);
}

static __device__ __forceinline__ float bcast0(float x) {
  return __builtin_bit_cast(float,
      __builtin_amdgcn_readfirstlane(__builtin_bit_cast(int, x)));
}

__global__ __launch_bounds__(32, 1)
void nssm_scan_kernel(const float* __restrict__ y,
    const float* __restrict__ We1, const float* __restrict__ be1,
    const float* __restrict__ We2, const float* __restrict__ be2,
    const float* __restrict__ Wt1, const float* __restrict__ bt1,
    const float* __restrict__ Wt2, const float* __restrict__ bt2,
    const float* __restrict__ Wd1, const float* __restrict__ bd1,
    const float* __restrict__ Wd2, const float* __restrict__ bd2,
    const float* __restrict__ Rm,
    float* __restrict__ out, int T)
{
  const int  lane = threadIdx.x & 31;
  const int  m    = lane & 15;
  const bool lo   = lane < 16;
  const int  jb   = lo ? 0 : 16;

  // ---------------- prologue (one-time) ----------------
  // Sinv = inv(R + 1e-5*I), 3x3 closed form (uniform across lanes).
  float a00=Rm[0]+1e-5f, a01=Rm[1],       a02=Rm[2];
  float a10=Rm[3],       a11=Rm[4]+1e-5f, a12=Rm[5];
  float a20=Rm[6],       a21=Rm[7],       a22=Rm[8]+1e-5f;
  float det = a00*(a11*a22-a12*a21) - a01*(a10*a22-a12*a20) + a02*(a10*a21-a11*a20);
  float id  = 1.f/det;
  float s00=(a11*a22-a12*a21)*id, s01=(a02*a21-a01*a22)*id, s02=(a01*a12-a02*a11)*id;
  float s10=(a12*a20-a10*a22)*id, s11=(a00*a22-a02*a20)*id, s12=(a02*a10-a00*a12)*id;
  float s20=(a10*a21-a11*a20)*id, s21=(a01*a20-a00*a21)*id, s22=(a00*a11-a01*a10)*id;

  const float bd2c0 = bd2[0], bd2c1 = bd2[1], bd2c2 = bd2[2];

  // Constant A-operand tiles (weights transposed; rows = output neurons in
  // slot order). f16 A 16x32 layout: lane l holds row l%16; K pairs:
  // l<16 -> {2r,2r+1 | r<8}, l>=16 -> +16 on K (contiguous K order).
  const int n0 = slot0(m);   // output column for tile0 rows
  const int n1 = slot1(m);   // output column for tile1 rows
  v16h A_u0, A_u1, A_wh0, A_wh1, A_lath, A_wf0, A_wf1, A_latf, A_y;
#pragma unroll
  for (int r = 0; r < 8; ++r) {
    const int k = 2*r + (lo ? 0 : 16);
#pragma unroll
    for (int e = 0; e < 2; ++e) {
      const int kk = k + e;   // K index == hidden slot == original unit id
      float wu0=0, wu1=0, wb0=0, wb1=0, wc0=0, wc1=0;
#pragma unroll
      for (int j = 0; j < 6; ++j) {
        const float t2 = Wt2[kk*6+j];
        const float e2 = We2[kk*6+j];
        wu0 += t2 * Wd1[j*32 + n0];        // (Wt2@Wd1)[kk][n0]
        wu1 += t2 * Wd1[j*32 + n1];
        wb0 += t2 * Wt1[j*32 + n0];        // (Wt2@Wt1)[kk][n0]
        wb1 += t2 * Wt1[j*32 + n1];
        wc0 += e2 * Wt1[j*32 + n0];        // (We2@Wt1)[kk][n0]
        wc1 += e2 * Wt1[j*32 + n1];
      }
      A_u0 [2*r+e] = (_Float16)wu0;
      A_u1 [2*r+e] = (_Float16)wu1;
      A_wh0[2*r+e] = (_Float16)wb0;
      A_wh1[2*r+e] = (_Float16)wb1;
      A_wf0[2*r+e] = (_Float16)wc0;
      A_wf1[2*r+e] = (_Float16)wc1;
      A_lath[2*r+e] = (_Float16)(m < 6 ? Wt2[kk*6+m] : 0.f);
      A_latf[2*r+e] = (_Float16)(m < 6 ? We2[kk*6+m] : 0.f);
      A_y  [2*r+e] = (_Float16)(m < 3 ? Wd2[kk*3+m] : 0.f);
    }
  }

  // Bias folded into the C operand (non-zero only in the N=0 column:
  // lanes 0 and 16). D layout: lane l holds N=l%16, M=r+(l<16?0:8).
  // Bias index must follow the same slot numbering as the A rows.
  v8f C_u0 = {}, C_u1 = {}, C_w0 = {}, C_w1 = {}, C_lat = {};
#pragma unroll
  for (int r = 0; r < 8; ++r) {
    const int row = (lo ? 0 : 8) + r;       // M within tile
    const int p0  = slot0(row);             // bias index for tile0
    const int p1  = slot1(row);             // bias index for tile1
    float bu0 = bd1[p0], bu1 = bd1[p1];
    float bw0 = bt1[p0], bw1 = bt1[p1];
#pragma unroll
    for (int j = 0; j < 6; ++j) {
      bu0 += bt2[j] * Wd1[j*32 + p0];
      bu1 += bt2[j] * Wd1[j*32 + p1];
      const float bb = bt2[j] + be2[j];
      bw0 += bb * Wt1[j*32 + p0];
      bw1 += bb * Wt1[j*32 + p1];
    }
    const float bl = (row < 6) ? (bt2[row] + be2[row]) : 0.f;
    const bool nn0 = (m == 0);
    C_u0[r] = nn0 ? bu0 : 0.f;
    C_u1[r] = nn0 ? bu1 : 0.f;
    C_w0[r] = nn0 ? bw0 : 0.f;
    C_w1[r] = nn0 ? bw1 : 0.f;
    C_lat[r] = nn0 ? bl : 0.f;
  }

  // Packed-f16 encoder layer-1 constants (per-lane, B-operand-direct;
  // f slots are identity).
  h2 wpk0[8], wpk1[8], wpk2[8], bpk[8];
#pragma unroll
  for (int r = 0; r < 8; ++r) {
    const int j = jb + 2*r;
    wpk0[r] = pk2(We1[j],      We1[j+1]);
    wpk1[r] = pk2(We1[32 + j], We1[32 + j + 1]);
    wpk2[r] = pk2(We1[64 + j], We1[64 + j + 1]);
    bpk[r]  = pk2(be1[j],      be1[j+1]);
  }

  // Initial recurrent hidden: h0 = relu(x0 @ Wt1 + bt1) = relu(bt1), x0 = 0.
  // (h slots are identity as well.)
  v16h Bh;
#pragma unroll
  for (int r = 0; r < 8; ++r) {
    const int j = jb + 2*r;
    Bh[2*r]   = (_Float16)fmaxf(bt1[j],   0.f);
    Bh[2*r+1] = (_Float16)fmaxf(bt1[j+1], 0.f);
  }

  // ---------------- sequential scan ----------------
  const v8f Cz = {};
  const h2  zh = {};
  float loss = 0.f;
  for (int t = 0; t < T; ++t) {
    // wave-uniform observation -> scalar loads
    const float yv0 = y[3*t], yv1 = y[3*t+1], yv2 = y[3*t+2];

    // h-pass: u, w(h part), latent(h part) -- biases arrive via C.
    v8f au0  = wmma(A_u0,  Bh, C_u0);
    v8f au1  = wmma(A_u1,  Bh, C_u1);
    v8f aw0  = wmma(A_wh0, Bh, C_w0);
    v8f aw1  = wmma(A_wh1, Bh, C_w1);
    v8f alat = wmma(A_lath, Bh, C_lat);

    // g = relu(u); yhat = g @ Wd2 (bd2 folded into innov below)
    v16h Bg = marshal_relu(au0, au1);
    v8f  ay = wmma(A_y, Bg, Cz);

    // yhat rows 0..2 live in lane 0 -> readfirstlane broadcast (scalar path)
    const float i0 = (yv0 - bd2c0) - bcast0(ay[0]);
    const float i1 = (yv1 - bd2c1) - bcast0(ay[1]);
    const float i2 = (yv2 - bd2c2) - bcast0(ay[2]);

    // loss += innov @ Sinv @ innov (uniform across lanes)
    const float q0 = i0*s00 + i1*s10 + i2*s20;
    const float q1 = i0*s01 + i1*s11 + i2*s21;
    const float q2 = i0*s02 + i1*s12 + i2*s22;
    loss += q0*i0 + q1*i1 + q2*i2;

    // f = relu(innov @ We1 + be1), produced directly in B-operand layout
    const h2 ip0 = pk2(i0, i0), ip1 = pk2(i1, i1), ip2 = pk2(i2, i2);
    v8i bf;
#pragma unroll
    for (int r = 0; r < 8; ++r) {
      h2 acc = bpk[r];
      acc = wpk0[r] * ip0 + acc;
      acc = wpk1[r] * ip1 + acc;
      acc = wpk2[r] * ip2 + acc;
      acc = __builtin_elementwise_max(acc, zh);
      bf[r] = __builtin_bit_cast(int, acc);
    }
    const v16h Bf = __builtin_bit_cast(v16h, bf);

    // f-pass accumulates into the same C/D chains: w += f@C', lat += f@We2'
    aw0  = wmma(A_wf0,  Bf, aw0);
    aw1  = wmma(A_wf1,  Bf, aw1);
    alat = wmma(A_latf, Bf, alat);

    // h' = relu(w) for the next step (again purely lane-local)
    Bh = marshal_relu(aw0, aw1);

    // latent x_new = rows 0..5 of alat, N=0 column -> lane 0 regs 0..5
    if (lane == 0) {
      float* o = out + 1 + (size_t)t * 6;
      o[0] = alat[0]; o[1] = alat[1]; o[2] = alat[2];
      o[3] = alat[3]; o[4] = alat[4]; o[5] = alat[5];
    }
  }

  if (lane == 0) out[0] = loss / (float)T;
}

extern "C" void kernel_launch(void* const* d_in, const int* in_sizes, int n_in,
                              void* d_out, int out_size, void* d_ws, size_t ws_size,
                              hipStream_t stream) {
  (void)n_in; (void)out_size; (void)d_ws; (void)ws_size;
  const float* y   = (const float*)d_in[0];
  const float* We1 = (const float*)d_in[1];
  const float* be1 = (const float*)d_in[2];
  const float* We2 = (const float*)d_in[3];
  const float* be2 = (const float*)d_in[4];
  const float* Wt1 = (const float*)d_in[5];
  const float* bt1 = (const float*)d_in[6];
  const float* Wt2 = (const float*)d_in[7];
  const float* bt2 = (const float*)d_in[8];
  const float* Wd1 = (const float*)d_in[9];
  const float* bd1 = (const float*)d_in[10];
  const float* Wd2 = (const float*)d_in[11];
  const float* bd2 = (const float*)d_in[12];
  // d_in[13] = Q (unused by the math, matches reference)
  const float* Rm  = (const float*)d_in[14];
  float* out = (float*)d_out;
  const int T = in_sizes[0] / 3;

  // Inherently sequential scan: one wave32 owns everything.
  nssm_scan_kernel<<<dim3(1), dim3(32), 0, stream>>>(
      y, We1, be1, We2, be2, Wt1, bt1, Wt2, bt2, Wd1, bd1, Wd2, bd2, Rm, out, T);
}
